// TopKRouter_28080496181687
// MI455X (gfx1250) — matrix operand order repaired and checked by reading, not compiled, not run on hardware
//
#include <hip/hip_runtime.h>
#include <hip/hip_bf16.h>

// ---------------- types ----------------
typedef __attribute__((ext_vector_type(16))) __bf16 v16bf;
typedef __attribute__((ext_vector_type(16), aligned(16))) __bf16 v16bfu; // 16B-aligned view for LDS
typedef __attribute__((ext_vector_type(8)))  float  v8f;
typedef __attribute__((ext_vector_type(4)))  float  v4f;

#define NTOK   16384
#define DMODEL 4096
#define NEXP   64
#define TOPK   2
#define CAP    640
#define NKSLOT (NTOK * TOPK)   // 32768 flat slots

#define KCHUNK 128             // K-extent staged in LDS per block iteration
#define LDSROW 136             // padded row stride (bf16 units): 272B -> bank-conflict-free

// ---------------------------------------------------------------------------
// Kernel 0: split W (f32) into bf16 hi/lo pair (bf16x3 emulated-f32 GEMM).
// ---------------------------------------------------------------------------
__global__ __launch_bounds__(256) void split_w(const float* __restrict__ W,
                                               __bf16* __restrict__ whi,
                                               __bf16* __restrict__ wlo) {
    int i = blockIdx.x * 256 + threadIdx.x;
    if (i < NEXP * DMODEL) {
        float v = W[i];
        __bf16 h = (__bf16)v;
        whi[i] = h;
        wlo[i] = (__bf16)(v - (float)h);
    }
}

// ---------------- A-fragment split helpers ----------------
__device__ __forceinline__ void cvt4(v4f a, v16bf& hi, v16bf& lo, int off) {
#pragma unroll
    for (int j = 0; j < 4; ++j) {
        float v = a[j];
        __bf16 h = (__bf16)v;
        hi[off + j] = h;
        lo[off + j] = (__bf16)(v - (float)h);
    }
}

// 16-bit A 16x32, contiguous k order used here: lane half 0 holds K 0..15, half 1 holds K 16..31.
__device__ __forceinline__ void loadA(const float* __restrict__ pa,
                                      v16bf& hi, v16bf& lo) {
    cvt4(*(const v4f*)(pa + 0),  hi, lo, 0);
    cvt4(*(const v4f*)(pa + 4),  hi, lo, 4);
    cvt4(*(const v4f*)(pa + 8),  hi, lo, 8);
    cvt4(*(const v4f*)(pa + 12), hi, lo, 12);
}

// ---------------------------------------------------------------------------
// Kernel 1: logits = x @ W^T + b via V_WMMA_F32_16X16X32_BF16 (bf16x3 split).
// Wave32: each wave computes TWO 16-token M-tiles (32 tokens) x 64 experts.
// W (the only reused operand) is staged per-block into LDS in KCHUNK slabs
// with CDNA5 async global->LDS DMA (no VGPR staging), cutting L2->WGP B
// traffic 8x (one copy feeds all 8 waves). x streams HBM->VGPR with prefetch.
// Block = 256 threads = 8 waves = 256 tokens. Grid = 64 blocks.
// ---------------------------------------------------------------------------
__global__ __launch_bounds__(256) void moe_logits_gemm(
        const float* __restrict__ x,
        const __bf16* __restrict__ whi,
        const __bf16* __restrict__ wlo,
        const float* __restrict__ bias,
        float* __restrict__ logits) {
    __shared__ __bf16 sh_hi[NEXP * LDSROW];   // 17 KB
    __shared__ __bf16 sh_lo[NEXP * LDSROW];   // 17 KB

    const int wave = threadIdx.x >> 5;
    const int lane = threadIdx.x & 31;
    const int half = lane >> 4;      // lane 16..31 -> second K-group
    const int l    = lane & 15;

    const int mbase = (blockIdx.x * 8 + wave) * 32;
    const float* __restrict__ xrow0 = x + (size_t)(mbase + l) * DMODEL;
    const float* __restrict__ xrow1 = x + (size_t)(mbase + 16 + l) * DMODEL;

    v8f acc0[4] = {};   // M-tile 0, 4 N-tiles of 16 experts
    v8f acc1[4] = {};   // M-tile 1

    for (int kc = 0; kc < DMODEL; kc += KCHUNK) {
        // All waves done reading the previous slab before the DMA overwrites it.
        __syncthreads();

        // Async-DMA this K-slab of W_hi/W_lo into LDS: 64 rows x 128 bf16 each.
        // 1024 b128 transfers per array; 4 per thread. Rows padded to LDSROW.
#pragma unroll
        for (int it = 0; it < 4; ++it) {
            int idx = it * 256 + threadIdx.x;        // 0..1023
            int e   = idx >> 4;                      // expert row
            int c8  = (idx & 15) * 8;                // bf16 column (8 per b128)
            unsigned lds_hi = (unsigned)(uintptr_t)&sh_hi[e * LDSROW + c8];
            unsigned lds_lo = (unsigned)(uintptr_t)&sh_lo[e * LDSROW + c8];
            unsigned long long g_hi =
                (unsigned long long)(uintptr_t)(whi + (size_t)e * DMODEL + kc + c8);
            unsigned long long g_lo =
                (unsigned long long)(uintptr_t)(wlo + (size_t)e * DMODEL + kc + c8);
            asm volatile("global_load_async_to_lds_b128 %0, %1, off"
                         :: "v"(lds_hi), "v"(g_hi) : "memory");
            asm volatile("global_load_async_to_lds_b128 %0, %1, off"
                         :: "v"(lds_lo), "v"(g_lo) : "memory");
        }
        asm volatile("s_wait_asynccnt 0x0" ::: "memory");
        __syncthreads();

        // Compute 4 K-steps (96 WMMAs) against the resident slab.
#pragma unroll
        for (int kkL = 0; kkL < KCHUNK; kkL += 32) {
            const int kk = kc + kkL;
            const float* pa0 = xrow0 + kk + half * 16;
            const float* pa1 = xrow1 + kk + half * 16;

            // Keep the one real HBM stream (x, 256 MB) ahead of the WMMA pipe.
            __builtin_prefetch(pa0 + 256, 0, 0);
            __builtin_prefetch(pa1 + 256, 0, 0);

            v16bf ahi0, alo0, ahi1, alo1;
            loadA(pa0, ahi0, alo0);
            loadA(pa1, ahi1, alo1);

#pragma unroll
            for (int nt = 0; nt < 4; ++nt) {
                // 16-bit B 32x16, contiguous k as in A: lane half selects K group of 16.
                // Padded LDS rows -> conflict-free ds_load_b128.
                const int roff = (nt * 16 + l) * LDSROW + kkL + half * 16;
                v16bf bhi = *(const v16bfu*)&sh_hi[roff];
                v16bf blo = *(const v16bfu*)&sh_lo[roff];

                // bf16x3: hi*hi + lo*hi + hi*lo, f32 accumulate; B reused by
                // both M-tiles.
                acc0[nt] = __builtin_amdgcn_wmma_f32_16x16x32_bf16(
                    false, ahi0, false, bhi, (short)0, acc0[nt], false, false);
                acc0[nt] = __builtin_amdgcn_wmma_f32_16x16x32_bf16(
                    false, alo0, false, bhi, (short)0, acc0[nt], false, false);
                acc0[nt] = __builtin_amdgcn_wmma_f32_16x16x32_bf16(
                    false, ahi0, false, blo, (short)0, acc0[nt], false, false);

                acc1[nt] = __builtin_amdgcn_wmma_f32_16x16x32_bf16(
                    false, ahi1, false, bhi, (short)0, acc1[nt], false, false);
                acc1[nt] = __builtin_amdgcn_wmma_f32_16x16x32_bf16(
                    false, alo1, false, bhi, (short)0, acc1[nt], false, false);
                acc1[nt] = __builtin_amdgcn_wmma_f32_16x16x32_bf16(
                    false, ahi1, false, blo, (short)0, acc1[nt], false, false);
            }
        }
    }

    // Epilogue: D layout — VGPR v: lanes 0-15 hold M=v, lanes 16-31 hold M=v+8.
#pragma unroll
    for (int nt = 0; nt < 4; ++nt) {
        float bb = bias[nt * 16 + l];
#pragma unroll
        for (int v = 0; v < 8; ++v) {
            int m = v + 8 * half;
            logits[(size_t)(mbase + m) * NEXP + nt * 16 + l]      = acc0[nt][v] + bb;
            logits[(size_t)(mbase + 16 + m) * NEXP + nt * 16 + l] = acc1[nt][v] + bb;
        }
    }
}

// ---------------------------------------------------------------------------
// Kernel 2: per-token softmax over 64 experts + top-2 (jax tie semantics:
// strict > keeps earliest index). Writes probs and flat (prob, eid) slots.
// ---------------------------------------------------------------------------
__global__ __launch_bounds__(256) void softmax_top2(
        const float* __restrict__ logits,
        float* __restrict__ probs,
        float* __restrict__ flat_probs,
        int*   __restrict__ flat_eids) {
    int t = blockIdx.x * 256 + threadIdx.x;
    if (t >= NTOK) return;
    const float* __restrict__ lr = logits + (size_t)t * NEXP;

    float vals[NEXP];
    float m = -3.402823466e38f;
#pragma unroll
    for (int i = 0; i < NEXP / 4; ++i) {
        v4f v = *(const v4f*)(lr + 4 * i);
#pragma unroll
        for (int j = 0; j < 4; ++j) {
            vals[4 * i + j] = v[j];
            m = fmaxf(m, v[j]);
        }
    }
    float s = 0.0f;
#pragma unroll
    for (int i = 0; i < NEXP; ++i) {
        float e = __expf(vals[i] - m);
        vals[i] = e;
        s += e;
    }
    float inv = 1.0f / s;

    float v1 = -1.0f, v2 = -1.0f;
    int   i1 = 0,     i2 = 0;
#pragma unroll
    for (int i = 0; i < NEXP; ++i) {
        float p = vals[i] * inv;
        probs[(size_t)t * NEXP + i] = p;
        if (p > v1)      { v2 = v1; i2 = i1; v1 = p; i1 = i; }
        else if (p > v2) { v2 = p;  i2 = i; }
    }
    flat_probs[2 * t + 0] = v1;  flat_eids[2 * t + 0] = i1;
    flat_probs[2 * t + 1] = v2;  flat_eids[2 * t + 1] = i2;
}

// ---------------------------------------------------------------------------
// Kernel 3: per-expert capacity-limited top-640, sorted descending.
// One block per expert. Keys: (prob_bits << 32) | ~token -> descending sort
// gives descending prob with ascending-token tie-break. Sentinel key 0
// unpacks to (prob=0.0, token=-1), exactly matching reference invalid slots.
// 64 regions x 256 KB = 16 MB of keys -> fully resident in 192 MB L2.
// ---------------------------------------------------------------------------
__global__ __launch_bounds__(1024) void expert_select(
        const float* __restrict__ flat_probs,
        const int*   __restrict__ flat_eids,
        unsigned long long* __restrict__ keys,
        float* __restrict__ eprob,
        int*   __restrict__ eidx) {
    const int e   = blockIdx.x;
    const int tid = threadIdx.x;
    unsigned long long* __restrict__ base = keys + (size_t)e * NKSLOT;

    // Build masked key array for this expert.
    for (int i = tid; i < NKSLOT; i += 1024) {
        unsigned long long k = 0ull;
        if (flat_eids[i] == e) {
            unsigned pb  = __float_as_uint(flat_probs[i]);       // prob >= 0
            unsigned tok = (unsigned)(i >> 1);                   // slot -> token
            k = ((unsigned long long)pb << 32) | (unsigned)(~tok);
        }
        base[i] = k;
    }
    __syncthreads();

    // Bitonic sort, descending, in L2-resident global workspace.
    for (int size = 2; size <= NKSLOT; size <<= 1) {
        for (int stride = size >> 1; stride > 0; stride >>= 1) {
            for (int i = tid; i < NKSLOT; i += 1024) {
                int j = i ^ stride;
                if (j > i) {
                    unsigned long long a = base[i];
                    unsigned long long b = base[j];
                    bool desc = ((i & size) == 0);
                    if ((a < b) == desc) { base[i] = b; base[j] = a; }
                }
            }
            __syncthreads();
        }
    }

    // Emit top-CAP.
    for (int c = tid; c < CAP; c += 1024) {
        unsigned long long k = base[c];
        eprob[(size_t)e * CAP + c] = __uint_as_float((unsigned)(k >> 32));
        eidx [(size_t)e * CAP + c] = (int)(~(unsigned)(k & 0xffffffffull));
    }
}

// ---------------------------------------------------------------------------
// Host launcher.
// ---------------------------------------------------------------------------
extern "C" void kernel_launch(void* const* d_in, const int* in_sizes, int n_in,
                              void* d_out, int out_size, void* d_ws, size_t ws_size,
                              hipStream_t stream) {
    const float* x = (const float*)d_in[0];
    const float* W = (const float*)d_in[1];
    const float* b = (const float*)d_in[2];
    // d_in[3] = k (==2), d_in[4] = expert_capacity (==640): compile-time here.

    // Output layout: logits | probs | expert_probs | expert_indices
    float* out    = (float*)d_out;
    float* logits = out;
    float* probs  = out + (size_t)NTOK * NEXP;
    float* eprob  = out + 2 * (size_t)NTOK * NEXP;
    int*   eidx   = (int*)(out + 2 * (size_t)NTOK * NEXP + (size_t)NEXP * CAP);

    // Workspace layout.
    char* ws = (char*)d_ws;
    __bf16* whi              = (__bf16*)ws;                               // 512 KB
    __bf16* wlo              = whi + (size_t)NEXP * DMODEL;               // 512 KB
    float*  flat_probs       = (float*)(ws + (1u << 20));                 // 128 KB
    int*    flat_eids        = (int*)(ws + (1u << 20) + (128u << 10));    // 128 KB
    unsigned long long* keys = (unsigned long long*)(ws + (1u << 20) + (256u << 10)); // 16 MB

    // 0) Split W into bf16 hi/lo.
    split_w<<<(NEXP * DMODEL + 255) / 256, 256, 0, stream>>>(W, whi, wlo);

    // 1) Logits GEMM (WMMA bf16x3, async-LDS W staging). 256 tokens per block.
    moe_logits_gemm<<<NTOK / 256, 256, 0, stream>>>(x, whi, wlo, b, logits);

    // 2) Softmax + top-2 routing.
    softmax_top2<<<(NTOK + 255) / 256, 256, 0, stream>>>(logits, probs,
                                                         flat_probs, flat_eids);

    // 3) Per-expert capacity-limited selection (sorted descending).
    expert_select<<<NEXP, 1024, 0, stream>>>(flat_probs, flat_eids, keys,
                                             eprob, eidx);
}
